// FeatureBatchSpatialGraphConv_34772055229048
// MI455X (gfx1250) — matrix operand order repaired and checked by reading, not compiled, not run on hardware
//
#include <hip/hip_runtime.h>
#include <hip/hip_bf16.h>

typedef __attribute__((ext_vector_type(16))) _Float16 v16h;
typedef __attribute__((ext_vector_type(8)))  float    v8f;

#define IN_C  128
#define OUT_C 128

// ---------------------------------------------------------------------------
// k0: zero the degree counters
// ---------------------------------------------------------------------------
__global__ void k_zero_deg(unsigned* __restrict__ deg, int n) {
    int i = blockIdx.x * blockDim.x + threadIdx.x;
    if (i < n) deg[i] = 0u;
}

// ---------------------------------------------------------------------------
// k1: degree count (symmetrized edges): deg[e0]++, deg[e1]++
// ---------------------------------------------------------------------------
__global__ void k_count_deg(const int* __restrict__ ei, unsigned* __restrict__ deg, int E) {
    int e = blockIdx.x * blockDim.x + threadIdx.x;
    if (e >= E) return;
    atomicAdd(&deg[ei[e]], 1u);
    atomicAdd(&deg[ei[E + e]], 1u);
}

// ---------------------------------------------------------------------------
// k2: dinv = rsqrt(deg + 1)   (self-loop weight 1 included in degree)
// ---------------------------------------------------------------------------
__global__ void k_dinv(const unsigned* __restrict__ deg, float* __restrict__ dinv, int n) {
    int i = blockIdx.x * blockDim.x + threadIdx.x;
    if (i < n) dinv[i] = rsqrtf((float)deg[i] + 1.0f);
}

// ---------------------------------------------------------------------------
// k3: xw = x @ W via v_wmma_f32_16x16x32_f16, f32 accumulation.
// Block = 256 threads = 8 waves. Wave w computes the 16x16 tile
// (M = blockIdx.x*16, N = w*16). K loop: 128 in 4 chunks of 32.
//
// Fragment layouts per CDNA5 ISA 7.12.2 (wave32):
//   A 16x32 f16 : lane l (l<16) -> M=l ; V0..3 pack K=0..7, V4..7 pack K=16..23
//                 lanes 16..31 -> same M, K offset +8.
//   B 32x16 f16 : lane l -> N=l%16 ; elements hold K in the same order as A.
//   C/D 16x16 f32: VGPR v -> M = v + (lane<16 ? 0 : 8); N = lane%16.
// ---------------------------------------------------------------------------
__global__ __launch_bounds__(256) void k_gemm_wmma(const float* __restrict__ x,
                                                   const float* __restrict__ W,
                                                   float* __restrict__ xw) {
    const int lane  = threadIdx.x & 31;
    const int n0    = (threadIdx.x >> 5) * 16;   // wave id -> N tile
    const int m0    = blockIdx.x * 16;
    const int hi8   = (lane & 16) ? 8 : 0;       // K offset for upper half-wave
    const float* xrow = x + (size_t)(m0 + (lane & 15)) * IN_C;

    v8f acc = {};
    for (int kc = 0; kc < IN_C; kc += 32) {
        // A fragment: 16 halfs for this lane
        v16h a;
#pragma unroll
        for (int h = 0; h < 8; ++h) {
            a[h]     = (_Float16)xrow[kc + hi8 + h];        // V0..3 : K = kc+hi8+0..7
            a[h + 8] = (_Float16)xrow[kc + 16 + hi8 + h];   // V4..7 : K = kc+16+hi8+0..7
        }
        // B fragment: lane -> N column n0+lane%16, 16 K values in same order as A
        const float* wrow = W + (size_t)kc * OUT_C + n0 + (lane & 15);
        v16h bm;
#pragma unroll
        for (int v = 0; v < 16; ++v) bm[v] = (_Float16)wrow[(size_t)(hi8 + (v & 7) + ((v & 8) ? 16 : 0)) * OUT_C];

        acc = __builtin_amdgcn_wmma_f32_16x16x32_f16(
            /*neg_a=*/false, a, /*neg_b=*/false, bm,
            /*c_mod=*/(short)0, acc, /*reuse_a=*/false, /*reuse_b=*/false);
    }

    // Store C tile: for VGPR v, lanes 0-15 write row m0+v cols n0..n0+15 (64B
    // contiguous), lanes 16-31 write row m0+8+v — coalesced.
    const int cn = n0 + (lane & 15);
#pragma unroll
    for (int v = 0; v < 8; ++v)
        xw[(size_t)(m0 + hi8 + v) * OUT_C + cn] = acc[v];
}

// ---------------------------------------------------------------------------
// k4: y = dinv^2 * xw + b   (self-loop contribution + bias, also zero-inits
// the accumulation target for the edge scatter)
// ---------------------------------------------------------------------------
__global__ void k_init_out(const float* __restrict__ xw, const float* __restrict__ dinv,
                           const float* __restrict__ b, float* __restrict__ y, int n) {
    int i = blockIdx.x * blockDim.x + threadIdx.x;   // over n * OUT_C
    if (i >= n * OUT_C) return;
    int node = i >> 7;       // /128
    int ch   = i & (OUT_C - 1);
    float di = dinv[node];
    y[i] = di * di * xw[i] + b[ch];
}

// ---------------------------------------------------------------------------
// k5: edge scatter. One wave32 per DIRECTED edge (2E total, symmetrized).
// Each lane handles 4 channels: float4 gather of xw[col] (L2-resident,
// 51.2MB < 192MB L2) then 4x global_atomic_add_f32 into y[row].
// ---------------------------------------------------------------------------
__global__ __launch_bounds__(256) void k_edge_scatter(const int* __restrict__ ei,
                                                      const float* __restrict__ dinv,
                                                      const float* __restrict__ xw,
                                                      float* __restrict__ y, int E) {
    int gtid = blockIdx.x * blockDim.x + threadIdx.x;
    int de   = gtid >> 5;          // directed edge id
    int lane = gtid & 31;
    if (de >= 2 * E) return;

    int r, c;
    if (de < E) { r = ei[de];          c = ei[E + de]; }
    else        { int e = de - E; r = ei[E + e]; c = ei[e]; }

    float norm = dinv[r] * dinv[c];

    const float4 v = ((const float4*)(xw + (size_t)c * OUT_C))[lane];
    float* yr = y + (size_t)r * OUT_C + lane * 4;
    atomicAdd(yr + 0, v.x * norm);
    atomicAdd(yr + 1, v.y * norm);
    atomicAdd(yr + 2, v.z * norm);
    atomicAdd(yr + 3, v.w * norm);
}

// ---------------------------------------------------------------------------
extern "C" void kernel_launch(void* const* d_in, const int* in_sizes, int n_in,
                              void* d_out, int out_size, void* d_ws, size_t ws_size,
                              hipStream_t stream) {
    const float* x  = (const float*)d_in[0];   // [N, 128]
    const float* W  = (const float*)d_in[1];   // [128, 128]
    const float* b  = (const float*)d_in[2];   // [128]
    const int*   ei = (const int*)d_in[3];     // [2, E] flat

    const int N = in_sizes[0] / IN_C;
    const int E = in_sizes[3] / 2;

    // Workspace layout (all 256B aligned):
    //   [deg: N u32][dinv: N f32][xw: N*128 f32]
    auto align256 = [](size_t v) { return (v + 255) & ~(size_t)255; };
    char* ws = (char*)d_ws;
    unsigned* deg  = (unsigned*)ws;
    float*    dinv = (float*)(ws + align256((size_t)N * 4));
    float*    xw   = (float*)(ws + align256((size_t)N * 4) + align256((size_t)N * 4));
    float*    y    = (float*)d_out;

    const int T = 256;

    // degree + normalization
    k_zero_deg  <<<(N + T - 1) / T, T, 0, stream>>>(deg, N);
    k_count_deg <<<(E + T - 1) / T, T, 0, stream>>>(ei, deg, E);
    k_dinv      <<<(N + T - 1) / T, T, 0, stream>>>(deg, dinv, N);

    // dense projection via WMMA: N is a multiple of 16 (100000 = 6250*16)
    k_gemm_wmma <<<N / 16, 256, 0, stream>>>(x, W, xw);

    // self-loop + bias init, then edge scatter-add
    int totalNC = N * OUT_C;
    k_init_out  <<<(totalNC + T - 1) / T, T, 0, stream>>>(xw, dinv, b, y, N);

    long long scatterThreads = (long long)2 * E * 32;
    int scatterBlocks = (int)((scatterThreads + T - 1) / T);
    k_edge_scatter <<<scatterBlocks, T, 0, stream>>>(ei, dinv, xw, y, E);
}
